// XunZi_M_GCN_79654463472115
// MI455X (gfx1250) — matrix operand order repaired and checked by reading, not compiled, not run on hardware
//
#include <hip/hip_runtime.h>
#include <hip/hip_bf16.h>
#include <stdint.h>

// ---------------------------------------------------------------------------
// MI455X (gfx1250) implementation of the GCN reference.
// Dense GEMMs via v_wmma_f32_16x16x32_bf16 (f32 accumulate), sparse
// aggregation via L2-resident global_atomic_add_f32.
// ---------------------------------------------------------------------------

#define N_GENE 40000
#define N_GOID 10000
#define N_NODE (N_GENE + N_GOID)
#define INPUT_DIM 1280
#define H1_DIM 256
#define H2_DIM 128
#define OUT_DIM 64
#define GOID_DIM 4096
#define DEC_H 1024

typedef __attribute__((ext_vector_type(16))) __bf16 v16bf;
typedef __attribute__((ext_vector_type(8)))  __bf16 v8bf;
typedef __attribute__((ext_vector_type(4)))  __bf16 v4bf;
typedef __attribute__((ext_vector_type(8)))  float  v8f;

// ---------------------------------------------------------------------------
// Tiled bf16 WMMA GEMM:  C[M,N] = op(A) @ B + bias, optional ReLU.
// Block tile: 64(M) x 128(N); 8 waves arranged 2x4; each wave owns a
// 32x32 output tile = 2x2 WMMA accumulators; K stepped by 32.
// A rows are gathered from two sources: row < split -> A0 (lda0),
// else A1 (row - split, lda1) -- fuses concat([gene_x, goid_t]).
//
// Staging is fully branch-free: out-of-range rows/cols are CLAMPED to the
// last valid row/col (valid memory, no masking). The garbage only lands in
// accumulator lanes that the (row<M, col<N)-guarded epilogue never stores.
// LDS is ping-pong double-buffered -> one barrier per K step, so next-tile
// global_load_b128 staging overlaps the WMMAs of the current tile.
// ---------------------------------------------------------------------------
template <int RELU>
__global__ __launch_bounds__(256)
void gemm_bf16_wmma(const float* __restrict__ A0, int lda0,
                    const float* __restrict__ A1, int lda1, int split,
                    const float* __restrict__ B,  int ldb,
                    const float* __restrict__ bias,
                    float* __restrict__ C, int ldc,
                    int M, int N, int K)
{
    __shared__ __bf16 aS[2][64][40];    // [buf][m][k], padded stride (80 B)
    __shared__ __bf16 bT[2][128][40];   // [buf][n][k] transposed, padded

    const int t    = threadIdx.x;
    const int wave = t >> 5;
    const int lane = t & 31;
    const int half = lane >> 4;
    const int l15  = lane & 15;
    const int wm   = wave >> 2;          // 0..1
    const int wn   = wave & 3;           // 0..3

    const int m0 = blockIdx.y * 64;
    const int n0 = blockIdx.x * 128;

    // ---- hoisted A staging pointers: 64 rows x 32 k = 512 float4 / block ----
    const float* ap[2];
    int am_[2], ak_[2];
    #pragma unroll
    for (int r = 0; r < 2; ++r) {
        const int i  = t + 256 * r;
        const int m  = i >> 3;
        const int kk = (i & 7) * 4;
        am_[r] = m; ak_[r] = kk;
        int gr = m0 + m;
        if (gr >= M) gr = M - 1;                       // clamp (branch-free load)
        ap[r] = ((gr < split) ? (A0 + (size_t)gr * lda0)
                              : (A1 + (size_t)(gr - split) * lda1)) + kk;
    }
    // ---- hoisted B staging pointers: 32 k x 128 n = 1024 float4 / block ----
    const float* bp[4];
    int bn_[4], bk_[4];
    #pragma unroll
    for (int r = 0; r < 4; ++r) {
        const int i  = t + 256 * r;
        const int kk = i >> 5;
        const int nq = i & 31;
        bn_[r] = nq * 4; bk_[r] = kk;
        int gn = n0 + nq * 4;
        if (gn >= N) gn = N - 4;                       // clamp (N % 4 == 0)
        bp[r] = B + (size_t)kk * ldb + gn;
    }

    v8f acc[2][2] = {};
    int buf = 0;

    for (int k0 = 0; k0 < K; k0 += 32, buf ^= 1) {
        // ---- stage A tile (global_load_b128 -> cvt -> ds_store_b64) ----
        #pragma unroll
        for (int r = 0; r < 2; ++r) {
            const float4 f = *(const float4*)(ap[r] + k0);
            __builtin_prefetch(ap[r] + k0 + 32, 0, 3);   // WGP-scope prefetch
            v4bf o;
            o[0] = (__bf16)f.x; o[1] = (__bf16)f.y;
            o[2] = (__bf16)f.z; o[3] = (__bf16)f.w;
            *(v4bf*)&aS[buf][am_[r]][ak_[r]] = o;
        }
        // ---- stage B tile transposed (global_load_b128 -> 4x ds_store_b16) --
        #pragma unroll
        for (int r = 0; r < 4; ++r) {
            const float4 f = *(const float4*)(bp[r] + (size_t)k0 * ldb);
            const int n = bn_[r], kk = bk_[r];
            bT[buf][n + 0][kk] = (__bf16)f.x;
            bT[buf][n + 1][kk] = (__bf16)f.y;
            bT[buf][n + 2][kk] = (__bf16)f.z;
            bT[buf][n + 3][kk] = (__bf16)f.w;
        }
        __syncthreads();   // single barrier per K step (ping-pong buffers)

        // ---- fragments via contiguous 16B LDS loads (ds_load_b128) ----
        union Frag { v16bf v; v8bf h[2]; };
        Frag a[2], b[2];
        #pragma unroll
        for (int im = 0; im < 2; ++im) {
            const int mrow = wm * 32 + im * 16 + l15;
            // A 16-bit layout: K = half*8 + e  and  K = 16 + half*8 + e
            a[im].h[0] = *(const v8bf*)&aS[buf][mrow][half * 8];
            a[im].h[1] = *(const v8bf*)&aS[buf][mrow][16 + half * 8];
        }
        #pragma unroll
        for (int in = 0; in < 2; ++in) {
            const int ncol = wn * 32 + in * 16 + l15;
            // B 16-bit layout: K = half*8 + e  and  K = 16 + half*8 + e
            b[in].h[0] = *(const v8bf*)&bT[buf][ncol][half * 8];
            b[in].h[1] = *(const v8bf*)&bT[buf][ncol][16 + half * 8];
        }
        #pragma unroll
        for (int im = 0; im < 2; ++im)
            #pragma unroll
            for (int in = 0; in < 2; ++in)
                acc[im][in] = __builtin_amdgcn_wmma_f32_16x16x32_bf16(
                    false, a[im].v, false, b[in].v,
                    (short)0, acc[im][in], false, false);
    }

    // ---- epilogue: C layout = (M = half*8 + j, N = lane&15) ----
    #pragma unroll
    for (int in = 0; in < 2; ++in) {
        const int col = n0 + wn * 32 + in * 16 + l15;
        if (col >= N) continue;
        const float bv = bias ? bias[col] : 0.0f;
        #pragma unroll
        for (int im = 0; im < 2; ++im) {
            const int row0 = m0 + wm * 32 + im * 16 + half * 8;
            #pragma unroll
            for (int j = 0; j < 8; ++j) {
                const int row = row0 + j;
                if (row < M) {
                    float v = acc[im][in][j] + bv;
                    if (RELU) v = fmaxf(v, 0.0f);
                    C[(size_t)row * ldc + col] = v;
                }
            }
        }
    }
}

// ---------------------------------------------------------------------------
// Sparse-aggregation helper kernels
// ---------------------------------------------------------------------------
__global__ void init_deg_kernel(float* deg, int n) {
    int i = blockIdx.x * blockDim.x + threadIdx.x;
    if (i < n) deg[i] = 1.0f;                    // self-loop contribution
}

__global__ void edge_deg_kernel(const int* __restrict__ dst, float* deg, int e) {
    int i = blockIdx.x * blockDim.x + threadIdx.x;
    if (i < e) atomicAdd(&deg[dst[i]], 1.0f);
}

__global__ void dinv_kernel(float* deg, int n) {
    int i = blockIdx.x * blockDim.x + threadIdx.x;
    if (i < n) {
        float d = deg[i];
        deg[i] = (d > 0.0f) ? rsqrtf(d) : 0.0f;
    }
}

// h[i,:] = xw[i,:] * dinv[i]^2   (self-loop term; also zero-fills workspace)
__global__ void selfloop_init_kernel(const float* __restrict__ xw,
                                     const float* __restrict__ dinv,
                                     float* __restrict__ h, int rows, int F) {
    size_t idx = (size_t)blockIdx.x * blockDim.x + threadIdx.x;
    size_t tot = (size_t)rows * F;
    if (idx < tot) {
        int i = (int)(idx / F);
        float d = dinv[i];
        h[idx] = xw[idx] * d * d;
    }
}

// one wave per edge: h[dst,:] += xw[src,:] * dinv[src]*dinv[dst]
__global__ __launch_bounds__(256)
void scatter_edges_kernel(const int* __restrict__ src, const int* __restrict__ dst,
                          const float* __restrict__ dinv,
                          const float* __restrict__ xw,
                          float* __restrict__ h, int e, int F) {
    int edge = blockIdx.x * (blockDim.x >> 5) + (threadIdx.x >> 5);
    int lane = threadIdx.x & 31;
    if (edge >= e) return;
    int s = src[edge], d = dst[edge];
    float nrm = dinv[s] * dinv[d];
    const float* xr = xw + (size_t)s * F;
    float*       hr = h  + (size_t)d * F;
    for (int c = lane; c < F; c += 32)
        atomicAdd(&hr[c], xr[c] * nrm);
}

// rows < limit: h = relu(h + bias)
__global__ void bias_relu_kernel(float* __restrict__ h, const float* __restrict__ bias,
                                 int rows, int F) {
    size_t idx = (size_t)blockIdx.x * blockDim.x + threadIdx.x;
    size_t tot = (size_t)rows * F;
    if (idx < tot)
        h[idx] = fmaxf(h[idx] + bias[idx % F], 0.0f);
}

__global__ void copy_kernel(const float* __restrict__ s, float* __restrict__ d, size_t n) {
    size_t idx = (size_t)blockIdx.x * blockDim.x + threadIdx.x;
    if (idx < n) d[idx] = s[idx];
}

// ---------------------------------------------------------------------------
// Host-side orchestration
// ---------------------------------------------------------------------------
static inline size_t align256(size_t x) { return (x + 255) & ~(size_t)255; }

extern "C" void kernel_launch(void* const* d_in, const int* in_sizes, int n_in,
                              void* d_out, int out_size, void* d_ws, size_t ws_size,
                              hipStream_t stream) {
    (void)n_in; (void)out_size; (void)ws_size;

    const float* x    = (const float*)d_in[0];   // (N, 4096)
    const int*   eidx = (const int*)  d_in[1];   // (2, E)
    const float* Wd1  = (const float*)d_in[3];
    const float* bd1  = (const float*)d_in[4];
    const float* Wd2  = (const float*)d_in[5];
    const float* bd2  = (const float*)d_in[6];
    const float* W1   = (const float*)d_in[7];
    const float* b1   = (const float*)d_in[8];
    const float* W2   = (const float*)d_in[9];
    const float* b2   = (const float*)d_in[10];
    const float* Wp1  = (const float*)d_in[11];
    const float* bp1  = (const float*)d_in[12];
    const float* Wp2  = (const float*)d_in[13];
    const float* bp2  = (const float*)d_in[14];
    const float* Wf   = (const float*)d_in[15];
    const float* bf   = (const float*)d_in[16];
    float* out = (float*)d_out;

    const int E = in_sizes[1] / 2;
    const int* src = eidx;
    const int* dst = eidx + E;

    // workspace layout
    char* ws = (char*)d_ws;
    size_t off = 0;
    float* deg    = (float*)(ws + off); off = align256(off + sizeof(float) * N_NODE);
    float* t1     = (float*)(ws + off); off = align256(off + sizeof(float) * (size_t)N_GOID * DEC_H);
    float* goid_t = (float*)(ws + off); off = align256(off + sizeof(float) * (size_t)N_GOID * INPUT_DIM);
    float* xw1    = (float*)(ws + off); off = align256(off + sizeof(float) * (size_t)N_NODE * H1_DIM);
    float* h1     = (float*)(ws + off); off = align256(off + sizeof(float) * (size_t)N_NODE * H1_DIM);
    float* p1     = (float*)(ws + off); off = align256(off + sizeof(float) * (size_t)N_GOID * H1_DIM);
    float* xw2    = (float*)(ws + off); off = align256(off + sizeof(float) * (size_t)N_NODE * H2_DIM);
    float* h2     = (float*)(ws + off); off = align256(off + sizeof(float) * (size_t)N_NODE * H2_DIM);

    const int BIG = 1 << 30;
    const float* goid_x = x + (size_t)N_GENE * GOID_DIM;

    // ---- degrees / symmetric norm ----
    init_deg_kernel<<<(N_NODE + 255) / 256, 256, 0, stream>>>(deg, N_NODE);
    edge_deg_kernel<<<(E + 255) / 256, 256, 0, stream>>>(dst, deg, E);
    dinv_kernel<<<(N_NODE + 255) / 256, 256, 0, stream>>>(deg, N_NODE);

    #define GEMM_GRID(M, N) dim3(((N) + 127) / 128, ((M) + 63) / 64)

    // ---- GOID decoder: t1 = relu(goid_x @ Wd1 + bd1) ----
    gemm_bf16_wmma<1><<<GEMM_GRID(N_GOID, DEC_H), 256, 0, stream>>>(
        goid_x, GOID_DIM, goid_x, GOID_DIM, BIG, Wd1, DEC_H, bd1,
        t1, DEC_H, N_GOID, DEC_H, GOID_DIM);
    // goid_t = relu(t1 @ Wd2 + bd2)
    gemm_bf16_wmma<1><<<GEMM_GRID(N_GOID, INPUT_DIM), 256, 0, stream>>>(
        t1, DEC_H, t1, DEC_H, BIG, Wd2, INPUT_DIM, bd2,
        goid_t, INPUT_DIM, N_GOID, INPUT_DIM, DEC_H);

    // ---- layer 1: xw1 = concat(gene_x, goid_t) @ W1 (fused gather) ----
    gemm_bf16_wmma<0><<<GEMM_GRID(N_NODE, H1_DIM), 256, 0, stream>>>(
        x, GOID_DIM, goid_t, INPUT_DIM, N_GENE, W1, H1_DIM, nullptr,
        xw1, H1_DIM, N_NODE, H1_DIM, INPUT_DIM);
    // p1 = goid_t @ Wp1 + bp1 (no relu)
    gemm_bf16_wmma<0><<<GEMM_GRID(N_GOID, H1_DIM), 256, 0, stream>>>(
        goid_t, INPUT_DIM, goid_t, INPUT_DIM, BIG, Wp1, H1_DIM, bp1,
        p1, H1_DIM, N_GOID, H1_DIM, INPUT_DIM);

    {   // aggregation 1
        size_t tot = (size_t)N_NODE * H1_DIM;
        selfloop_init_kernel<<<(unsigned)((tot + 255) / 256), 256, 0, stream>>>(
            xw1, deg, h1, N_NODE, H1_DIM);
        scatter_edges_kernel<<<(E + 7) / 8, 256, 0, stream>>>(
            src, dst, deg, xw1, h1, E, H1_DIM);
        size_t gtot = (size_t)N_GENE * H1_DIM;
        bias_relu_kernel<<<(unsigned)((gtot + 255) / 256), 256, 0, stream>>>(
            h1, b1, N_GENE, H1_DIM);
        size_t ptot = (size_t)N_GOID * H1_DIM;   // h1[N_GENE:] = p1 (raw)
        copy_kernel<<<(unsigned)((ptot + 255) / 256), 256, 0, stream>>>(
            p1, h1 + (size_t)N_GENE * H1_DIM, ptot);
    }

    // ---- layer 2: xw2 = h1 @ W2 ----
    gemm_bf16_wmma<0><<<GEMM_GRID(N_NODE, H2_DIM), 256, 0, stream>>>(
        h1, H1_DIM, h1, H1_DIM, BIG, W2, H2_DIM, nullptr,
        xw2, H2_DIM, N_NODE, H2_DIM, H1_DIM);

    {   // aggregation 2
        size_t tot = (size_t)N_NODE * H2_DIM;
        selfloop_init_kernel<<<(unsigned)((tot + 255) / 256), 256, 0, stream>>>(
            xw2, deg, h2, N_NODE, H2_DIM);
        scatter_edges_kernel<<<(E + 7) / 8, 256, 0, stream>>>(
            src, dst, deg, xw2, h2, E, H2_DIM);
        size_t gtot = (size_t)N_GENE * H2_DIM;
        bias_relu_kernel<<<(unsigned)((gtot + 255) / 256), 256, 0, stream>>>(
            h2, b2, N_GENE, H2_DIM);
    }
    // h2[N_GENE:] = p1 @ Wp2 + bp2  (GEMM writes directly into h2 rows)
    gemm_bf16_wmma<0><<<GEMM_GRID(N_GOID, H2_DIM), 256, 0, stream>>>(
        p1, H1_DIM, p1, H1_DIM, BIG, Wp2, H2_DIM, bp2,
        h2 + (size_t)N_GENE * H2_DIM, H2_DIM, N_GOID, H2_DIM, H1_DIM);

    // ---- output: out = h2 @ Wf + bf ----
    gemm_bf16_wmma<0><<<GEMM_GRID(N_NODE, OUT_DIM), 256, 0, stream>>>(
        h2, H2_DIM, h2, H2_DIM, BIG, Wf, OUT_DIM, bf,
        out, OUT_DIM, N_NODE, OUT_DIM, H2_DIM);

    #undef GEMM_GRID
}
